// HybridNKATModel_38585986187664
// MI455X (gfx1250) — matrix operand order repaired and checked by reading, not compiled, not run on hardware
//
#include <hip/hip_runtime.h>

// ---------------------------------------------------------------------------
// HybridNKATModel forward for MI455X (gfx1250).
//
// Each KAN layer:   out[b,j] = gelu( sum_{i,k} exp(-2(tanh(h)[b,i]-knot_k)^2)
//                                    * coeffs[i,j,k] * scale[i,j] + bias[j] )
// knots are broadcast over (i,j)  ->  layer == GEMM  Phi[B,Din*16] x W[Din*16,Dout]
// with W[(i,k),j] = coeffs[i,j,k]*scale[i,j].  M=32 (two 16-row WMMA tiles),
// K in {64,8192,4096,2048} (all %32==0), N = Dout (padded to 16).
// B operand is read straight from global f32 coeffs (each used exactly once:
// weight-bandwidth bound problem) and converted to f16 in registers.
// Column padding handled branch-free: clamped index + zeroed scale.
// ---------------------------------------------------------------------------

typedef __attribute__((ext_vector_type(16))) _Float16 v16h;
typedef __attribute__((ext_vector_type(8)))  _Float16 v8h;
typedef __attribute__((ext_vector_type(8)))  float    v8f;

#define BATCH 32
#define GRID_ 16

__device__ __forceinline__ float gelu_exact(float x) {
    return 0.5f * x * (1.0f + erff(x * 0.70710678118654752f));
}

// phi[b, i*16+k] = exp(-2*(tanh(h[b,i]) - knots[i,0,k])^2)  (f16, row-major [B,K])
__global__ void kan_basis_kernel(const float* __restrict__ h,
                                 const float* __restrict__ knots,
                                 _Float16* __restrict__ phi,
                                 int Din, int Dout) {
    const int K = Din * GRID_;
    int t = blockIdx.x * blockDim.x + threadIdx.x;
    if (t >= BATCH * K) return;
    int b = t / K;
    int q = t - b * K;
    int i = q >> 4;
    int k = q & 15;
    float a  = tanhf(h[b * Din + i]);
    float kn = knots[(size_t)i * Dout * GRID_ + k];   // knots broadcast over j
    float d  = a - kn;
    phi[(size_t)b * K + q] = (_Float16)expf(-2.0f * d * d);
}

// One wave per 16-column tile; computes all 32 batch rows (2 WMMA accumulators).
__global__ void __launch_bounds__(32)
kan_gemm_kernel(const _Float16* __restrict__ phi,   // [BATCH, K] f16 row-major
                const float* __restrict__ coeffs,   // [Din, Dout, 16] f32
                const float* __restrict__ scale,    // [Din, Dout] f32
                const float* __restrict__ bias,     // [Dout] f32
                float* __restrict__ out,            // [BATCH, Dout] f32
                int Din, int Dout) {
    const int K    = Din * GRID_;
    const int lane = threadIdx.x;
    const int n    = lane & 15;
    const int hi   = lane >> 4;
    const int j    = blockIdx.x * 16 + n;
    // Branch-free column padding: clamp index so loads stay in-bounds,
    // zero the scale so padded lanes contribute exact zeros to WMMA-B.
    const int   jc  = (j < Dout) ? j : (Dout - 1);
    const float msk = (j < Dout) ? 1.0f : 0.0f;

    const _Float16* prow0 = phi + (size_t)(lane & 15) * K + hi * 16;        // rows 0..15
    const _Float16* prow1 = phi + (size_t)((lane & 15) + 16) * K + hi * 16; // rows 16..31

    v8f acc0 = {};
    v8f acc1 = {};

    for (int kk = 0; kk < K; kk += 32) {
        // ---- A tiles (16x32 f16): lane holds row (lane&15);
        //      elems 0..15 = K[kk+hi*16 .. kk+hi*16+15]
        v8h lo0 = *(const v8h*)(prow0 + kk);
        v8h hh0 = *(const v8h*)(prow0 + kk + 8);
        v8h lo1 = *(const v8h*)(prow1 + kk);
        v8h hh1 = *(const v8h*)(prow1 + kk + 8);
        v16h a0 = __builtin_shufflevector(lo0, hh0, 0,1,2,3,4,5,6,7,8,9,10,11,12,13,14,15);
        v16h a1 = __builtin_shufflevector(lo1, hh1, 0,1,2,3,4,5,6,7,8,9,10,11,12,13,14,15);

        // ---- B tile (32x16 f16): lane holds column j, K = kk+hi*16 .. +15.
        //      That K-slice lies inside one grid block: coeffs[i, j, 0..15] * scale[i,j].
        const int i0 = (kk + hi * 16) >> 4;
        const float* wp = coeffs + ((size_t)i0 * Dout + jc) * GRID_;
        const float  sc = scale[(size_t)i0 * Dout + jc] * msk;

        v16h bh;
#pragma unroll
        for (int v = 0; v < 16; ++v) bh[v] = (_Float16)(wp[v] * sc);

        if (kk + 32 < K)  // uniform branch: prefetch next step's coeff line (i += 2)
            __builtin_prefetch(coeffs + ((size_t)(i0 + 2) * Dout + jc) * GRID_, 0, 1);

        acc0 = __builtin_amdgcn_wmma_f32_16x16x32_f16(false, a0, false, bh,
                                                      (short)0, acc0, false, false);
        acc1 = __builtin_amdgcn_wmma_f32_16x16x32_f16(false, a1, false, bh,
                                                      (short)0, acc1, false, false);
    }

    if (j < Dout) {
        float bb = bias[j];
#pragma unroll
        for (int r = 0; r < 8; ++r) {
            int m = hi * 8 + r;                       // C layout: m = hi*8+r, n = lane&15
            out[(size_t)m * Dout + j]        = gelu_exact(acc0[r] + bb);
            out[(size_t)(m + 16) * Dout + j] = gelu_exact(acc1[r] + bb);
        }
    }
}

// theta MLP [1,128,64,32,1]: tiny, one block of 128 threads, loop over batch.
__global__ void __launch_bounds__(128)
theta_kernel(const float* __restrict__ es,           // [32,1], positive
             const float* __restrict__ thetaBaseLog, // [1]
             const float* __restrict__ w1, const float* __restrict__ nb1,
             const float* __restrict__ w2, const float* __restrict__ nb2,
             const float* __restrict__ w3, const float* __restrict__ nb3,
             const float* __restrict__ w4, const float* __restrict__ nb4,
             float* __restrict__ theta_out) {        // [32]
    __shared__ float h1[128];
    __shared__ float h2[64];
    __shared__ float h3[32];
    const int t = threadIdx.x;
    const float tbl = thetaBaseLog[0];
    for (int b = 0; b < BATCH; ++b) {
        float r0 = log10f(es[b]);
        h1[t] = gelu_exact(r0 * w1[t] + nb1[t]);
        __syncthreads();
        if (t < 64) {
            float acc = nb2[t];
            for (int i = 0; i < 128; ++i) acc += h1[i] * w2[i * 64 + t];
            h2[t] = gelu_exact(acc);
        }
        __syncthreads();
        if (t < 32) {
            float acc = nb3[t];
            for (int i = 0; i < 64; ++i) acc += h2[i] * w3[i * 32 + t];
            h3[t] = gelu_exact(acc);
        }
        __syncthreads();
        if (t == 0) {
            float acc = nb4[0];
            for (int i = 0; i < 32; ++i) acc += h3[i] * w4[i];
            theta_out[b] = expf(tbl + acc);
        }
        __syncthreads();
    }
}

// dirac_field complex64 [32,4,1]: interleave (real, 0) into d_out[0..255]
__global__ void __launch_bounds__(128)
pack_dirac_kernel(const float* __restrict__ h, float* __restrict__ out) {
    int t = threadIdx.x;      // 128 = 32*4
    out[2 * t]     = h[t];
    out[2 * t + 1] = 0.0f;
}

extern "C" void kernel_launch(void* const* d_in, const int* in_sizes, int n_in,
                              void* d_out, int out_size, void* d_ws, size_t ws_size,
                              hipStream_t stream) {
    // Input order: x, energy_scale, 4x(coeffs,knots,scale,bias), theta_base_log,
    //              4x(w,b)
    const float* x   = (const float*)d_in[0];
    const float* es  = (const float*)d_in[1];
    const float* kc[4]; const float* kk[4]; const float* ks[4]; const float* kb[4];
    for (int l = 0; l < 4; ++l) {
        kc[l] = (const float*)d_in[2 + 4 * l + 0];
        kk[l] = (const float*)d_in[2 + 4 * l + 1];
        ks[l] = (const float*)d_in[2 + 4 * l + 2];
        kb[l] = (const float*)d_in[2 + 4 * l + 3];
    }
    const float* tbl = (const float*)d_in[18];
    const float* nw[4]; const float* nb[4];
    for (int l = 0; l < 4; ++l) {
        nw[l] = (const float*)d_in[19 + 2 * l + 0];
        nb[l] = (const float*)d_in[19 + 2 * l + 1];
    }

    // Workspace: two f32 activation buffers (<=32*512) + f16 phi (<=32*8192)
    char* ws = (char*)d_ws;
    float*    hA  = (float*)(ws);
    float*    hB  = (float*)(ws + 64 * 1024);
    _Float16* phi = (_Float16*)(ws + 128 * 1024);

    const int   Din[4]  = {4, 512, 256, 128};
    const int   Dout[4] = {512, 256, 128, 4};
    const float* lin[4] = {x, hA, hB, hA};
    float*      lout[4] = {hA, hB, hA, hB};

    for (int l = 0; l < 4; ++l) {
        const int K  = Din[l] * GRID_;
        const int nb_basis = (BATCH * K + 255) / 256;
        kan_basis_kernel<<<nb_basis, 256, 0, stream>>>(lin[l], kk[l], phi,
                                                       Din[l], Dout[l]);
        const int ntiles = (Dout[l] + 15) / 16;
        kan_gemm_kernel<<<ntiles, 32, 0, stream>>>(phi, kc[l], ks[l], kb[l],
                                                   lout[l], Din[l], Dout[l]);
    }

    float* out = (float*)d_out;
    pack_dirac_kernel<<<1, 128, 0, stream>>>(hB, out);                 // [0..255]
    theta_kernel<<<1, 128, 0, stream>>>(es, tbl,
                                        nw[0], nb[0], nw[1], nb[1],
                                        nw[2], nb[2], nw[3], nb[3],
                                        out + 2 * BATCH * 4);          // [256..287]
}
